// GCNModel_15779709846111
// MI455X (gfx1250) — compile-verified, matrix operand order checked by reading, not run on hardware
//
#include <hip/hip_runtime.h>
#include <hip/hip_bf16.h>

typedef __attribute__((ext_vector_type(16))) _Float16 v16h;
typedef __attribute__((ext_vector_type(8)))  float    v8f;

#define HDIM 128

// ---------------------------------------------------------------------------
// Degree / normalization
// ---------------------------------------------------------------------------
__global__ void k_init_deg(float* deg, int N) {
    int i = blockIdx.x * blockDim.x + threadIdx.x;
    if (i < N) deg[i] = 1.0f;               // self-loop contributes 1
}

__global__ void k_add_deg(const long long* __restrict__ dst, float* deg, int E) {
    int i = blockIdx.x * blockDim.x + threadIdx.x;
    if (i < E) atomicAdd(&deg[dst[i]], 1.0f);
}

__global__ void k_dinv(float* deg, int N) {
    int i = blockIdx.x * blockDim.x + threadIdx.x;
    if (i < N) deg[i] = rsqrtf(deg[i]);     // deg >= 1 always (self loops)
}

// ---------------------------------------------------------------------------
// Dense GEMM  Y[N,128] = X[N,128] @ W[128,128]   (f16 WMMA, f32 accumulate)
// block = 256 threads (8 waves) computes a 64-row x 128-col slab.
// wave w: rows (w>>1)*16 .. +15, cols (w&1)*64 .. +63  -> 4 accumulators,
// 16 WMMAs per wave, A fragment reused across 4 column tiles.
// Fused epilogue: Y = acc, Yself = acc * dinv[row]^2  (self-loop init for
// the subsequent atomic edge aggregation).
// ---------------------------------------------------------------------------
__global__ void k_gemm_wmma(const float* __restrict__ X,
                            const float* __restrict__ W,
                            const float* __restrict__ dinv,
                            float* __restrict__ Y,
                            float* __restrict__ Yself,
                            int Nrows) {
    __shared__ _Float16 lA[64 * HDIM];    // A tile row-major [r][k]   (16 KB)
    __shared__ _Float16 lBt[HDIM * HDIM]; // full B transposed [n][k]  (32 KB)

    const int tid     = threadIdx.x;
    const int rowBase = blockIdx.x * 64;

    // stage A (64x128), convert f32 -> f16
    for (int i = tid; i < 64 * HDIM; i += 256) {
        int r = i >> 7, k = i & 127;
        int gr = rowBase + r;
        lA[i] = (_Float16)((gr < Nrows) ? X[(size_t)gr * HDIM + k] : 0.0f);
    }
    // stage full B transposed: W is [k][n] row-major; store Bt[n][k]
    for (int i = tid; i < HDIM * HDIM; i += 256) {
        int n = i >> 7, k = i & 127;
        lBt[n * HDIM + k] = (_Float16)W[(size_t)k * HDIM + n];
    }
    __syncthreads();

    const int wave    = tid >> 5;
    const int lane    = tid & 31;
    const int waveRow = wave >> 1;            // 0..3
    const int colBase = (wave & 1) * 64;      // 0 or 64
    const int rowTile = rowBase + waveRow * 16;
    if (rowTile >= Nrows) return;             // wave-uniform exit

    const int grp = lane >> 4;                // half-wave group (0/1)
    const int l15 = lane & 15;

    const unsigned* pA = (const unsigned*)&lA[(waveRow * 16 + l15) * HDIM];

    v8f acc0 = {}, acc1 = {}, acc2 = {}, acc3 = {};
    #pragma unroll
    for (int kb = 0; kb < HDIM; kb += 32) {
        // 16-bit A layout: lane grp g, VGPR v<4  -> K = kb + 8g + 2v
        //                            VGPR v>=4 -> K = kb + 16 + 8g + 2(v-4)
        int base = (kb >> 1) + (grp << 2);    // b32 index into the LDS row
        v16h a;
        unsigned* au = (unsigned*)&a;
        #pragma unroll
        for (int v = 0; v < 4; ++v) {
            au[v]     = pA[base + v];
            au[v + 4] = pA[base + 8 + v];
        }
        #pragma unroll
        for (int c = 0; c < 4; ++c) {
            const unsigned* pB =
                (const unsigned*)&lBt[(colBase + c * 16 + l15) * HDIM];
            v16h b;
            unsigned* bu = (unsigned*)&b;
            #pragma unroll
            for (int v = 0; v < 4; ++v) {
                bu[v]     = pB[base + v];
                bu[v + 4] = pB[base + 8 + v];
            }
            v8f& acc = (c == 0) ? acc0 : (c == 1) ? acc1 : (c == 2) ? acc2 : acc3;
            acc = __builtin_amdgcn_wmma_f32_16x16x32_f16(
                      false, a, false, b, (short)0, acc, false, false);
        }
    }

    // Epilogue: D layout lane = N + 16*(M>=8), vgpr v = M%8
    #pragma unroll
    for (int c = 0; c < 4; ++c) {
        const v8f& acc = (c == 0) ? acc0 : (c == 1) ? acc1 : (c == 2) ? acc2 : acc3;
        #pragma unroll
        for (int v = 0; v < 8; ++v) {
            int row = rowTile + v + 8 * grp;
            if (row < Nrows) {
                size_t off = (size_t)row * HDIM + colBase + c * 16 + l15;
                float val = acc[v];
                float di  = dinv[row];
                Y[off]     = val;
                Yself[off] = val * di * di;   // self-loop contribution
            }
        }
    }
}

// ---------------------------------------------------------------------------
// Edge aggregation: wave per edge, lane carries float4 (32*4 = 128 features)
// out[dst] += h[src] * dinv[src]*dinv[dst]      (L2-resident atomics)
// ---------------------------------------------------------------------------
__global__ void k_edge_agg(const long long* __restrict__ src,
                           const long long* __restrict__ dst,
                           const float* __restrict__ dinv,
                           const float* __restrict__ h,
                           float* __restrict__ out, int E) {
    int gt   = blockIdx.x * blockDim.x + threadIdx.x;
    int wid  = gt >> 5;
    int lane = gt & 31;
    if (wid >= E) return;
    long long s = src[wid];
    long long d = dst[wid];
    float nrm = dinv[s] * dinv[d];
    const float4* hs = (const float4*)(h + (size_t)s * HDIM);
    float4 v = hs[lane];
    float* o = out + (size_t)d * HDIM + lane * 4;
    atomicAdd(o + 0, v.x * nrm);
    atomicAdd(o + 1, v.y * nrm);
    atomicAdd(o + 2, v.z * nrm);
    atomicAdd(o + 3, v.w * nrm);
}

// ---------------------------------------------------------------------------
// Fused bias + eval-BatchNorm + ReLU (in place)
// y = relu(g * (x + b) * 1/sqrt(1+eps) + be)
// ---------------------------------------------------------------------------
__global__ void k_bn_relu(float* __restrict__ h,
                          const float* __restrict__ bias,
                          const float* __restrict__ g,
                          const float* __restrict__ be, int total) {
    const float scale = 0.99999500003749974f;   // 1/sqrt(1 + 1e-5)
    int i = blockIdx.x * blockDim.x + threadIdx.x;
    if (i < total) {
        int c = i & 127;
        float v = g[c] * (h[i] + bias[c]) * scale + be[c];
        h[i] = v > 0.0f ? v : 0.0f;
    }
}

// ---------------------------------------------------------------------------
// Global mean pool
// ---------------------------------------------------------------------------
__global__ void k_zero(float* p, int n) {
    int i = blockIdx.x * blockDim.x + threadIdx.x;
    if (i < n) p[i] = 0.0f;
}

__global__ void k_pool_sum(const float* __restrict__ h,
                           const long long* __restrict__ batch,
                           float* __restrict__ sums,
                           float* __restrict__ cnts, int N) {
    int gt   = blockIdx.x * blockDim.x + threadIdx.x;
    int wid  = gt >> 5;
    int lane = gt & 31;
    if (wid >= N) return;
    long long g = batch[wid];
    const float4* hv = (const float4*)(h + (size_t)wid * HDIM);
    float4 v = hv[lane];
    float* s = sums + (size_t)g * HDIM + lane * 4;
    atomicAdd(s + 0, v.x);
    atomicAdd(s + 1, v.y);
    atomicAdd(s + 2, v.z);
    atomicAdd(s + 3, v.w);
    if (lane == 0) atomicAdd(&cnts[g], 1.0f);
}

__global__ void k_pool_div(float* __restrict__ sums,
                           const float* __restrict__ cnts, int G) {
    int i = blockIdx.x * blockDim.x + threadIdx.x;
    if (i < G * HDIM) sums[i] /= fmaxf(cnts[i >> 7], 1.0f);
}

// ---------------------------------------------------------------------------
// Classifier: per-graph block, 128 -> 64 (ReLU) -> 10
// ---------------------------------------------------------------------------
__global__ void k_classify(const float* __restrict__ pooled,
                           const float* __restrict__ Wc1,
                           const float* __restrict__ bc1,
                           const float* __restrict__ Wc2,
                           const float* __restrict__ bc2,
                           float* __restrict__ out, int G) {
    __shared__ float p[HDIM];
    __shared__ float z[64];
    int g = blockIdx.x;
    if (g >= G) return;
    int t = threadIdx.x;
    p[t] = pooled[(size_t)g * HDIM + t];
    __syncthreads();
    if (t < 64) {
        float acc = bc1[t];
        #pragma unroll 8
        for (int k = 0; k < HDIM; ++k) acc += p[k] * Wc1[k * 64 + t];
        z[t] = acc > 0.0f ? acc : 0.0f;
    }
    __syncthreads();
    if (t < 10) {
        float acc = bc2[t];
        #pragma unroll
        for (int k = 0; k < 64; ++k) acc += z[k] * Wc2[k * 10 + t];
        out[(size_t)g * 10 + t] = acc;
    }
}

// ---------------------------------------------------------------------------
extern "C" void kernel_launch(void* const* d_in, const int* in_sizes, int n_in,
                              void* d_out, int out_size, void* d_ws, size_t ws_size,
                              hipStream_t stream) {
    const float*     x    = (const float*)d_in[0];
    const long long* ei   = (const long long*)d_in[1];
    const long long* batch= (const long long*)d_in[2];
    const float*     W[3] = {(const float*)d_in[3], (const float*)d_in[5], (const float*)d_in[7]};
    const float*     b[3] = {(const float*)d_in[4], (const float*)d_in[6], (const float*)d_in[8]};
    const float*     g[3] = {(const float*)d_in[9],  (const float*)d_in[11], (const float*)d_in[13]};
    const float*     be[3]= {(const float*)d_in[10], (const float*)d_in[12], (const float*)d_in[14]};
    const float*     Wc1  = (const float*)d_in[15];
    const float*     bc1  = (const float*)d_in[16];
    const float*     Wc2  = (const float*)d_in[17];
    const float*     bc2  = (const float*)d_in[18];
    float* out = (float*)d_out;

    const int N = in_sizes[2];          // batch has N elements
    const int E = in_sizes[1] / 2;      // edge_index is (2, E)
    const int G = out_size / 10;

    // workspace layout (floats)
    float* ws   = (float*)d_ws;
    float* hA   = ws;                                   // N*128
    float* hB   = hA + (size_t)N * HDIM;                // N*128
    float* dinv = hB + (size_t)N * HDIM;                // N
    float* sums = dinv + N;                             // G*128
    float* cnts = sums + (size_t)G * HDIM;              // G

    const long long* src = ei;
    const long long* dst = ei + E;

    const int B256 = 256;
    auto cdiv = [](long long a, long long bq) { return (int)((a + bq - 1) / bq); };

    // 1) degree -> dinv
    k_init_deg<<<cdiv(N, B256), B256, 0, stream>>>(dinv, N);
    k_add_deg <<<cdiv(E, B256), B256, 0, stream>>>(dst, dinv, E);
    k_dinv    <<<cdiv(N, B256), B256, 0, stream>>>(dinv, N);

    // 2) three GCN layers
    const float* cur = x;
    for (int l = 0; l < 3; ++l) {
        // GEMM + fused self-loop init:  hA = cur@W,  hB = hA * dinv^2
        k_gemm_wmma<<<cdiv(N, 64), 256, 0, stream>>>(cur, W[l], dinv, hA, hB, N);
        // edge atomics into hB
        k_edge_agg <<<cdiv((long long)E * 32, B256), B256, 0, stream>>>(src, dst, dinv, hA, hB, E);
        // bias + BN + ReLU in place
        k_bn_relu  <<<cdiv((long long)N * HDIM, B256), B256, 0, stream>>>(hB, b[l], g[l], be[l], N * HDIM);
        cur = hB;
    }

    // 3) global mean pool
    k_zero    <<<cdiv((long long)G * HDIM, B256), B256, 0, stream>>>(sums, G * HDIM);
    k_zero    <<<cdiv(G, B256), B256, 0, stream>>>(cnts, G);
    k_pool_sum<<<cdiv((long long)N * 32, B256), B256, 0, stream>>>(hB, batch, sums, cnts, N);
    k_pool_div<<<cdiv((long long)G * HDIM, B256), B256, 0, stream>>>(sums, cnts, G);

    // 4) classifier MLP
    k_classify<<<G, HDIM, 0, stream>>>(sums, Wc1, bc1, Wc2, bc2, out, G);
}
